// GATBackbone_69784628625940
// MI455X (gfx1250) — matrix operand order repaired and checked by reading, not compiled, not run on hardware
//
#include <hip/hip_runtime.h>
#include <math.h>

// ---------------- CDNA5 WMMA types ----------------
typedef __attribute__((ext_vector_type(16))) __bf16 v16bf;
typedef __attribute__((ext_vector_type(8)))  float  v8f;

#define HID   256
#define HEADS 4
#define NEGS  0.2f
#define BNEPS 1e-5f

// ---------------- utility kernels ----------------
__global__ void gat_fill_f32(float* __restrict__ p, float v, int n) {
    int t = blockIdx.x * blockDim.x + threadIdx.x;
    if (t < n) p[t] = v;
}

__global__ void gat_cvt_bf16(const float* __restrict__ x, __bf16* __restrict__ y, int n) {
    int t = blockIdx.x * blockDim.x + threadIdx.x;
    if (t < n) y[t] = (__bf16)x[t];
}

// Pack W [K x Nout] (row-major fp32) into per-lane WMMA B-fragment order (bf16).
// Memory layout: ((ntile*nkb + kb)*32 + lane)*16 + j16, i.e. each lane owns 32
// contiguous bytes per (ntile,kb) so the GEMM loads one aligned v16bf per lane.
// Fragment mapping (16-bit B 32x16): element s holds K = (lane>>4)*8 + s (s<8), 16 + (lane>>4)*8 + (s-8) (s>=8), N = lane&15.
__global__ void gat_pack_w(const float* __restrict__ W, __bf16* __restrict__ Wpk,
                           int K, int Nout) {
    int t = blockIdx.x * blockDim.x + threadIdx.x;
    int total = K * Nout;
    if (t >= total) return;
    int nkb   = K >> 5;
    int j16   = t & 15;
    int rest  = t >> 4;
    int lane  = rest & 31;
    int rest2 = rest >> 5;
    int kb    = rest2 % nkb;
    int ntile = rest2 / nkb;
    int v = j16 >> 1, j = j16 & 1;
    int kidx = kb * 32 + (j16 < 8 ? (lane >> 4) * 8 + 2 * v + j : 16 + (lane >> 4) * 8 + 2 * v + j - 8);
    int nidx = ntile * 16 + (lane & 15);
    Wpk[t] = (__bf16)W[(size_t)kidx * Nout + nidx];
}

// ---------------- bf16 WMMA GEMM: C[M,Nout] = A[M,K] * B[K,Nout] + bias ----------------
// A: bf16 row-major. Bpk: pre-packed fragments. One wave per 16x16 C tile,
// 4 waves per block. K multiple of 32; M multiple of 16; Nout multiple of 16.
__global__ void __launch_bounds__(128)
gat_gemm_wmma(const __bf16* __restrict__ A, const __bf16* __restrict__ Bpk,
              const float* __restrict__ bias, float* __restrict__ C,
              int M, int Nout, int K) {
    const int lane  = threadIdx.x & 31;
    const int wave  = threadIdx.x >> 5;
    const int mtile = blockIdx.y * 4 + wave;
    const int ntile = blockIdx.x;
    const int mtiles = M >> 4;
    if (mtile >= mtiles) return;

    const int nkb   = K >> 5;
    const int row   = mtile * 16 + (lane & 15);   // A: M index (both lane halves)
    const int khalf = lane >> 4;                  // A: K-half selector

    v8f acc = {};
    const __bf16* pArow = A + (size_t)row * K + khalf * 8;
    const v16bf* pB = (const v16bf*)Bpk + (size_t)ntile * nkb * 32 + lane;

    for (int kb = 0; kb < nkb; ++kb) {
        const __bf16* pA = pArow + kb * 32;
        __builtin_prefetch(pA + 32, 0, 0);        // global_prefetch_b8 next A chunk
        v16bf a;
#pragma unroll
        for (int i = 0; i < 8; ++i) a[i] = pA[i];       // VGPR0..3: K = khalf*8 + 0..7
#pragma unroll
        for (int i = 0; i < 8; ++i) a[8 + i] = pA[16 + i]; // VGPR4..7: K = 16 + khalf*8 + 0..7
        v16bf b = pB[(size_t)kb * 32];
        acc = __builtin_amdgcn_wmma_f32_16x16x32_bf16(
                  false, a, false, b, (short)0, acc, false, false);
    }

    const int n  = ntile * 16 + (lane & 15);
    const int m0 = mtile * 16 + khalf * 8;
    const float bv = bias[n];
#pragma unroll
    for (int r = 0; r < 8; ++r)
        C[(size_t)(m0 + r) * Nout + n] = acc[r] + bv;
}

// ---------------- float atomic max (segment_max) ----------------
__device__ __forceinline__ void atomicMaxF(float* addr, float v) {
    if (v >= 0.0f) atomicMax((int*)addr, __float_as_int(v));
    else           atomicMin((unsigned int*)addr, __float_as_uint(v));
}

// ---------------- edge attention scores: e = att . leaky_relu(xl[src]+xr[dst]) ----------------
// One wave32 per edge; lanes stride the D dimension per head. Also atomic-max into mmax[dst].
__global__ void __launch_bounds__(256)
gat_edge_scores(const float* __restrict__ xl, const float* __restrict__ xr,
                const int* __restrict__ src, const int* __restrict__ dst,
                const float* __restrict__ att, float* __restrict__ escore,
                float* __restrict__ mmax, int E, int HD, int D) {
    __shared__ float s_att[1024];
    for (int i = threadIdx.x; i < HD; i += blockDim.x) s_att[i] = att[i];
    __syncthreads();

    const int widx = blockIdx.x * 8 + (threadIdx.x >> 5);
    if (widx >= E) return;
    const int lane = threadIdx.x & 31;
    const int sn = src[widx], dn = dst[widx];
    const float* pl = xl + (size_t)sn * HD;
    const float* pr = xr + (size_t)dn * HD;

#pragma unroll
    for (int h = 0; h < HEADS; ++h) {
        float part = 0.0f;
        for (int d = lane; d < D; d += 32) {
            float v = pl[h * D + d] + pr[h * D + d];
            v = (v > 0.0f) ? v : NEGS * v;
            part += v * s_att[h * D + d];
        }
#pragma unroll
        for (int m = 16; m > 0; m >>= 1) part += __shfl_xor(part, m, 32);
        if (lane == 0) {
            escore[(size_t)widx * HEADS + h] = part;
            atomicMaxF(&mmax[dn * HEADS + h], part);
        }
    }
}

// ---------------- softmax numerator + denominator ----------------
__global__ void gat_edge_exp(float* __restrict__ escore, const int* __restrict__ dst,
                             const float* __restrict__ mmax, float* __restrict__ denom,
                             int EH) {
    int t = blockIdx.x * blockDim.x + threadIdx.x;
    if (t >= EH) return;
    int eidx = t >> 2, h = t & 3;
    int dn = dst[eidx];
    float a = __expf(escore[t] - mmax[dn * HEADS + h]);
    escore[t] = a;
    atomicAdd(&denom[dn * HEADS + h], a);
}

// ---------------- weighted message scatter: accum[dst] += xl[src] * alpha ----------------
__global__ void __launch_bounds__(256)
gat_edge_scatter(const float* __restrict__ xl, const float* __restrict__ escore,
                 const float* __restrict__ denom, const int* __restrict__ src,
                 const int* __restrict__ dst, float* __restrict__ accum,
                 int E, int HD, int D) {
    const int widx = blockIdx.x * 8 + (threadIdx.x >> 5);
    if (widx >= E) return;
    const int lane = threadIdx.x & 31;
    const int sn = src[widx], dn = dst[widx];
    float alpha[HEADS];
#pragma unroll
    for (int h = 0; h < HEADS; ++h)
        alpha[h] = escore[(size_t)widx * HEADS + h] /
                   (denom[dn * HEADS + h] + 1e-16f);
    const float* pl = xl + (size_t)sn * HD;
    float* po = accum + (size_t)dn * HD;
#pragma unroll
    for (int h = 0; h < HEADS; ++h)
        for (int d = lane; d < D; d += 32)
            atomicAdd(&po[h * D + d], pl[h * D + d] * alpha[h]);
}

// ---------------- bias (+head-mean for last layer) + BN channel statistics ----------------
// 256 channels; thread = channel, blocks stride rows; per-thread partials -> 2 atomics each.
__global__ void __launch_bounds__(256)
gat_post_bias_stats(const float* __restrict__ accum, const float* __restrict__ bias,
                    float* __restrict__ hbuf, float* __restrict__ stats,
                    int N, int HD, int concat) {
    const int c = threadIdx.x;
    const float bv = bias[c];
    float s = 0.0f, s2 = 0.0f;
    for (int n = blockIdx.x; n < N; n += gridDim.x) {
        float h;
        if (concat) {
            h = accum[(size_t)n * HD + c] + bv;
        } else {
            const float* p = accum + (size_t)n * HD;
            h = 0.25f * (p[c] + p[256 + c] + p[512 + c] + p[768 + c]) + bv;
        }
        hbuf[(size_t)n * HID + c] = h;
        s += h; s2 += h * h;
    }
    atomicAdd(&stats[c], s);
    atomicAdd(&stats[HID + c], s2);
}

// ---------------- batchnorm + ELU ----------------
__global__ void gat_bn_elu(const float* __restrict__ hbuf, const float* __restrict__ stats,
                           const float* __restrict__ gamma, const float* __restrict__ beta,
                           float* __restrict__ out, int N) {
    int t = blockIdx.x * blockDim.x + threadIdx.x;
    if (t >= N * HID) return;
    int c = t & (HID - 1);
    float inv_n = 1.0f / (float)N;
    float mu  = stats[c] * inv_n;
    float var = stats[HID + c] * inv_n - mu * mu;
    float y = gamma[c] * (hbuf[t] - mu) * rsqrtf(var + BNEPS) + beta[c];
    out[t] = (y > 0.0f) ? y : expm1f(y);
}

// ---------------- driver ----------------
extern "C" void kernel_launch(void* const* d_in, const int* in_sizes, int n_in,
                              void* d_out, int out_size, void* d_ws, size_t ws_size,
                              hipStream_t stream) {
    const int N = in_sizes[0] / HID;         // 50000
    const int E = in_sizes[1] / 2;           // 400000
    const float* x0 = (const float*)d_in[0];
    const int* ei   = (const int*)d_in[1];
    const int* src  = ei;
    const int* dst  = ei + E;

    // carve workspace (assumes ws_size sufficient; ~750 MB)
    char* w = (char*)d_ws;
    size_t off = 0;
    auto carve = [&](size_t bytes) -> char* {
        char* p = w + off;
        off = (off + bytes + 255) & ~(size_t)255;
        return p;
    };
    __bf16* xbf   = (__bf16*)carve((size_t)N * HID * sizeof(__bf16));
    __bf16* wlpk  = (__bf16*)carve((size_t)HID * 1024 * sizeof(__bf16));
    __bf16* wrpk  = (__bf16*)carve((size_t)HID * 1024 * sizeof(__bf16));
    float*  xl    = (float*)carve((size_t)N * 1024 * sizeof(float));
    float*  xr    = (float*)carve((size_t)N * 1024 * sizeof(float));
    float*  accum = (float*)carve((size_t)N * 1024 * sizeof(float));
    float*  esc   = (float*)carve((size_t)E * HEADS * sizeof(float));
    float*  mmax  = (float*)carve((size_t)N * HEADS * sizeof(float));
    float*  denom = (float*)carve((size_t)N * HEADS * sizeof(float));
    float*  hbuf  = (float*)carve((size_t)N * HID * sizeof(float));
    float*  xcur  = (float*)carve((size_t)N * HID * sizeof(float));
    float*  stats = (float*)carve(2 * HID * sizeof(float));

    const int mtiles = N >> 4;
    const int eblocks = (E + 7) / 8;

    for (int i = 0; i < 3; ++i) {
        const int HD = (i < 2) ? HID : 1024;
        const int D  = HD / HEADS;
        const float* Wl   = (const float*)d_in[2 + i * 8 + 0];
        const float* bl   = (const float*)d_in[2 + i * 8 + 1];
        const float* Wr   = (const float*)d_in[2 + i * 8 + 2];
        const float* br   = (const float*)d_in[2 + i * 8 + 3];
        const float* att  = (const float*)d_in[2 + i * 8 + 4];
        const float* bias = (const float*)d_in[2 + i * 8 + 5];
        const float* gmm  = (const float*)d_in[2 + i * 8 + 6];
        const float* bta  = (const float*)d_in[2 + i * 8 + 7];
        const float* xin  = (i == 0) ? x0 : xcur;
        float* xout = (i == 2) ? (float*)d_out : xcur;

        // 1. x -> bf16
        {
            int n = N * HID;
            gat_cvt_bf16<<<(n + 255) / 256, 256, 0, stream>>>(xin, xbf, n);
        }
        // 2. pack weights into WMMA B-fragment order
        {
            int n = HID * HD;
            gat_pack_w<<<(n + 255) / 256, 256, 0, stream>>>(Wl, wlpk, HID, HD);
            gat_pack_w<<<(n + 255) / 256, 256, 0, stream>>>(Wr, wrpk, HID, HD);
        }
        // 3. WMMA GEMMs: xl = x@Wl + bl ; xr = x@Wr + br
        {
            dim3 grid(HD / 16, (mtiles + 3) / 4);
            gat_gemm_wmma<<<grid, 128, 0, stream>>>(xbf, wlpk, bl, xl, N, HD, HID);
            gat_gemm_wmma<<<grid, 128, 0, stream>>>(xbf, wrpk, br, xr, N, HD, HID);
        }
        // 4. init segment buffers
        {
            int nh = N * HEADS;
            gat_fill_f32<<<(nh + 255) / 256, 256, 0, stream>>>(mmax, -__builtin_inff(), nh);
            gat_fill_f32<<<(nh + 255) / 256, 256, 0, stream>>>(denom, 0.0f, nh);
            int na = N * HD;
            gat_fill_f32<<<(na + 255) / 256, 256, 0, stream>>>(accum, 0.0f, na);
            gat_fill_f32<<<(2 * HID + 255) / 256, 256, 0, stream>>>(stats, 0.0f, 2 * HID);
        }
        // 5. attention scores + segment max
        gat_edge_scores<<<eblocks, 256, 0, stream>>>(xl, xr, src, dst, att, esc, mmax, E, HD, D);
        // 6. exp + segment sum
        {
            int eh = E * HEADS;
            gat_edge_exp<<<(eh + 255) / 256, 256, 0, stream>>>(esc, dst, mmax, denom, eh);
        }
        // 7. message scatter
        gat_edge_scatter<<<eblocks, 256, 0, stream>>>(xl, esc, denom, src, dst, accum, E, HD, D);
        // 8. bias (+mean for last layer) + BN statistics
        gat_post_bias_stats<<<256, 256, 0, stream>>>(accum, bias, hbuf, stats, N, HD, (i < 2) ? 1 : 0);
        // 9. batchnorm + elu
        {
            int n = N * HID;
            gat_bn_elu<<<(n + 255) / 256, 256, 0, stream>>>(hbuf, stats, gmm, bta, xout, N);
        }
    }
}
